// NetVLAD_29446295781941
// MI455X (gfx1250) — matrix operand order repaired and checked by reading, not compiled, not run on hardware
//
#include <hip/hip_runtime.h>
#include <hip/hip_bf16.h>

// ---------------------------------------------------------------------------
// NetVLAD for MI455X (gfx1250): bf16 WMMA GEMMs with fp32 accumulate.
//  - 64-wide K/p chunks: 8 v_wmma per barrier interval, fragments preloaded
//  - softmax reductions via ds_swizzle_b32 (group-of-32 XOR mode)
//  - fused a_sum*C into the VLAD store, LDS-resident normalization
// ---------------------------------------------------------------------------

#define EPSF 1e-12f

typedef __bf16  bf16_t;
typedef bf16_t  v16bf __attribute__((ext_vector_type(16)));
typedef float   v8f   __attribute__((ext_vector_type(8)));
typedef unsigned int u32;
typedef u32     u32x4 __attribute__((ext_vector_type(4)));

union FragU { v16bf v; u32x4 q[2]; };

__device__ __forceinline__ unsigned short f2bf_raw(float f) {
    u32 u = __builtin_bit_cast(u32, f);
    u += 0x7FFFu + ((u >> 16) & 1u);          // round to nearest even
    return (unsigned short)(u >> 16);
}
__device__ __forceinline__ u32 pack2bf(float a, float b) {
    return (u32)f2bf_raw(a) | ((u32)f2bf_raw(b) << 16);
}

// ds_swizzle_b32, group-of-32 XOR mode: offset[14:10]=xor, [9:5]=or, [4:0]=and
template <int XORMASK>
__device__ __forceinline__ float swz_xor(float v) {
    int i = __builtin_bit_cast(int, v);
    i = __builtin_amdgcn_ds_swizzle(i, ((XORMASK & 31) << 10) | 0x1F);
    return __builtin_bit_cast(float, i);
}

// Problem constants
#define NIMG 64
#define HW   1024
#define DCH  512
#define KCL  64
#define PTOT (NIMG * HW)

// =====================================================================
// Kernel 1: a = softmax(x @ W + b) -> a_ws (bf16 [PTOT][64]); a_sum[n][k]
// 256 threads / 8 waves; block = 128 pixels x 64 cols; D chunk 64.
// =====================================================================
__global__ __launch_bounds__(256)
void k_assign_softmax(const float* __restrict__ x,
                      const float* __restrict__ W,
                      const float* __restrict__ bias,
                      unsigned short* __restrict__ a_ws,
                      float* __restrict__ a_sum)
{
    __shared__ __attribute__((aligned(16))) unsigned short xs[128 * 64];   // x chunk bf16 [row][64]
    __shared__ __attribute__((aligned(16))) unsigned short wsT[64 * 72];   // W^T chunk [k][64], 144B rows
    __shared__ float bias_s[64];
    __shared__ float asum_s[64];

    const int tid  = threadIdx.x;
    const int wave = tid >> 5;
    const int lane = tid & 31;
    const int ln   = lane & 15;
    const int half = lane >> 4;

    const int p0 = blockIdx.x * 128;
    const int n  = p0 >> 10;

    if (tid < 64) { bias_s[tid] = bias[tid]; asum_s[tid] = 0.0f; }

    v8f acc[4] = {};

    for (int d0 = 0; d0 < DCH; d0 += 64) {
        // --- stage x[p0..p0+127][d0..d0+63] -> bf16 LDS, packed 8B stores ---
        {
            const int row = tid >> 1;
            const int c0  = (tid & 1) * 32;
            const float* src = x + (size_t)(p0 + row) * DCH + d0 + c0;
            #pragma unroll
            for (int i = 0; i < 8; ++i) {
                float4 f = *(const float4*)(src + 4 * i);
                uint2 pk; pk.x = pack2bf(f.x, f.y); pk.y = pack2bf(f.z, f.w);
                *(uint2*)&xs[row * 64 + c0 + 4 * i] = pk;
            }
        }
        // --- stage W[d0..d0+63][0..63] transposed -> wsT[k][d] ---
        {
            const int d  = tid >> 2;              // 0..63
            const int kb = (tid & 3) * 16;
            const float* src = W + (size_t)(d0 + d) * KCL + kb;
            float tmp[16];
            #pragma unroll
            for (int i = 0; i < 4; ++i) *(float4*)(tmp + 4 * i) = *(const float4*)(src + 4 * i);
            #pragma unroll
            for (int j = 0; j < 16; ++j)
                wsT[(kb + j) * 72 + d] = f2bf_raw(tmp[j]);
        }
        __syncthreads();

        // --- preload all fragments, then dense WMMA clause (8 per interval) ---
        FragU af[2];
        #pragma unroll
        for (int s = 0; s < 2; ++s) {
            const int base = (16 * wave + ln) * 64 + s * 32 + half * 8;
            af[s].q[0] = *(const u32x4*)&xs[base];
            af[s].q[1] = *(const u32x4*)&xs[base + 16];
        }
        FragU bf[2][4];
        #pragma unroll
        for (int s = 0; s < 2; ++s)
            #pragma unroll
            for (int t = 0; t < 4; ++t) {
                const int base = (16 * t + ln) * 72 + s * 32 + 8 * half;
                bf[s][t].q[0] = *(const u32x4*)&wsT[base];
                bf[s][t].q[1] = *(const u32x4*)&wsT[base + 16];
            }
        #pragma unroll
        for (int s = 0; s < 2; ++s)
            #pragma unroll
            for (int t = 0; t < 4; ++t)
                acc[t] = __builtin_amdgcn_wmma_f32_16x16x32_bf16(
                    false, af[s].v, false, bf[s][t].v, (short)0, acc[t], false, false);
        __syncthreads();
    }

    // --- bias add ---
    #pragma unroll
    for (int t = 0; t < 4; ++t) {
        const float bc = bias_s[16 * t + ln];
        #pragma unroll
        for (int vv = 0; vv < 8; ++vv) acc[t][vv] += bc;
    }

    // --- softmax over K=64 per row; ds_swizzle XOR reductions in 16-lane halves ---
    #pragma unroll
    for (int vv = 0; vv < 8; ++vv) {
        float m = fmaxf(fmaxf(acc[0][vv], acc[1][vv]), fmaxf(acc[2][vv], acc[3][vv]));
        m = fmaxf(m, swz_xor<1>(m));
        m = fmaxf(m, swz_xor<2>(m));
        m = fmaxf(m, swz_xor<4>(m));
        m = fmaxf(m, swz_xor<8>(m));
        float s = 0.0f;
        #pragma unroll
        for (int t = 0; t < 4; ++t) {
            float e = __expf(acc[t][vv] - m);
            acc[t][vv] = e;
            s += e;
        }
        s += swz_xor<1>(s);
        s += swz_xor<2>(s);
        s += swz_xor<4>(s);
        s += swz_xor<8>(s);
        const float inv = 1.0f / s;
        #pragma unroll
        for (int t = 0; t < 4; ++t) acc[t][vv] *= inv;
    }

    // --- column sums (LDS ds_add_f32), write a (bf16) ---
    #pragma unroll
    for (int t = 0; t < 4; ++t) {
        float cs = 0.0f;
        #pragma unroll
        for (int vv = 0; vv < 8; ++vv) cs += acc[t][vv];
        cs += swz_xor<16>(cs);                    // combine the two half-waves
        if (lane < 16) atomicAdd(&asum_s[16 * t + lane], cs);

        const int col = 16 * t + ln;
        #pragma unroll
        for (int vv = 0; vv < 8; ++vv) {
            const int row = p0 + 16 * wave + vv + 8 * half;
            a_ws[(size_t)row * KCL + col] = f2bf_raw(acc[t][vv]);
        }
    }
    __syncthreads();
    if (tid < 64) atomicAdd(&a_sum[n * KCL + tid], asum_s[tid]);
}

// =====================================================================
// Kernel 2: v[n][d][k] = sum_p x[n,p,d]*a[n,p,k] + a_sum[n,k]*C[d,k]
// 256 blocks = (n, 128-row d-block); p chunk 64 -> 8 WMMA per interval.
// =====================================================================
__global__ __launch_bounds__(256)
void k_vlad_gemm(const float* __restrict__ x,
                 const unsigned short* __restrict__ a_ws,
                 const float* __restrict__ a_sum,
                 const float* __restrict__ Cmat,
                 float* __restrict__ out)
{
    __shared__ __attribute__((aligned(16))) unsigned short xsT[128 * 72];  // x^T chunk [d][64p], 144B rows
    __shared__ __attribute__((aligned(16))) unsigned short asT[64 * 72];   // a^T chunk [k][64p]
    __shared__ float asum_s[64];

    const int tid  = threadIdx.x;
    const int wave = tid >> 5;
    const int lane = tid & 31;
    const int ln   = lane & 15;
    const int half = lane >> 4;

    const int n   = blockIdx.x >> 2;
    const int d0  = (blockIdx.x & 3) * 128;
    const int pgB = n * HW;

    if (tid < 64) asum_s[tid] = a_sum[n * KCL + tid];

    v8f acc[4] = {};

    for (int pl0 = 0; pl0 < HW; pl0 += 64) {
        const int pg = pgB + pl0;
        if (pl0 + 64 < HW)
            __builtin_prefetch(x + (size_t)(pg + 64 + (tid >> 2)) * DCH + d0 + (tid & 3) * 32, 0, 0);

        // --- stage x[pg..pg+63][d0..d0+127] transposed -> xsT[d][p] ---
        {
            const int pl = tid >> 2;              // 0..63
            const int dl = (tid & 3) * 32;
            const float* src = x + (size_t)(pg + pl) * DCH + d0 + dl;
            float tmp[32];
            #pragma unroll
            for (int i = 0; i < 8; ++i) *(float4*)(tmp + 4 * i) = *(const float4*)(src + 4 * i);
            #pragma unroll
            for (int j = 0; j < 32; ++j)
                xsT[(dl + j) * 72 + pl] = f2bf_raw(tmp[j]);
        }
        // --- stage a[pg..pg+63][0..63] transposed -> asT[k][p] ---
        {
            const int pl = tid >> 2;              // 0..63
            const int kb = (tid & 3) * 16;
            const u32x4 a0 = *(const u32x4*)(a_ws + (size_t)(pg + pl) * KCL + kb);
            const u32x4 a1 = *(const u32x4*)(a_ws + (size_t)(pg + pl) * KCL + kb + 8);
            #pragma unroll
            for (int j = 0; j < 8; ++j)
                asT[(kb + j) * 72 + pl] = (unsigned short)(a0[j >> 1] >> ((j & 1) * 16));
            #pragma unroll
            for (int j = 0; j < 8; ++j)
                asT[(kb + 8 + j) * 72 + pl] = (unsigned short)(a1[j >> 1] >> ((j & 1) * 16));
        }
        __syncthreads();

        FragU af[2];
        #pragma unroll
        for (int s = 0; s < 2; ++s) {
            const int base = (16 * wave + ln) * 72 + s * 32 + half * 8;
            af[s].q[0] = *(const u32x4*)&xsT[base];
            af[s].q[1] = *(const u32x4*)&xsT[base + 16];
        }
        FragU bf[2][4];
        #pragma unroll
        for (int s = 0; s < 2; ++s)
            #pragma unroll
            for (int t = 0; t < 4; ++t) {
                const int base = (16 * t + ln) * 72 + s * 32 + 8 * half;
                bf[s][t].q[0] = *(const u32x4*)&asT[base];
                bf[s][t].q[1] = *(const u32x4*)&asT[base + 16];
            }
        #pragma unroll
        for (int s = 0; s < 2; ++s)
            #pragma unroll
            for (int t = 0; t < 4; ++t)
                acc[t] = __builtin_amdgcn_wmma_f32_16x16x32_bf16(
                    false, af[s].v, false, bf[s][t].v, (short)0, acc[t], false, false);
        __syncthreads();
    }

    // --- store with fused  + a_sum[n,k] * C[d,k] ---
    #pragma unroll
    for (int t = 0; t < 4; ++t) {
        const int k  = 16 * t + ln;
        const float as = asum_s[k];
        #pragma unroll
        for (int vv = 0; vv < 8; ++vv) {
            const int d = d0 + 16 * wave + vv + 8 * half;
            const float cv = Cmat[(size_t)d * KCL + k];
            out[((size_t)n * DCH + d) * KCL + k] = acc[t][vv] + as * cv;
        }
    }
}

// =====================================================================
// Kernel 3: per-image normalization, LDS-resident (139KB < 320KB/WGP).
// =====================================================================
__global__ __launch_bounds__(256)
void k_normalize(float* __restrict__ v)
{
    extern __shared__ float smem[];
    float* vb   = smem;                 // 512*68
    float* red  = smem + 512 * 68;      // 256 partials
    float* scal = red + 256;            // 64
    float* gsp  = scal + 64;            // 1

    const int tid = threadIdx.x;
    float* vn = v + (size_t)blockIdx.x * DCH * KCL;

    #pragma unroll 4
    for (int pass = 0; pass < 32; ++pass) {
        const int idx = pass * 256 + tid;
        const int d = idx >> 4, k4 = (idx & 15) * 4;
        *(float4*)(vb + d * 68 + k4) = *(const float4*)(vn + d * 64 + k4);
    }
    __syncthreads();

    {
        const int k = tid & 63, q = tid >> 6;
        float a = 0.0f;
        const float* col = vb + (q * 128) * 68 + k;
        #pragma unroll 8
        for (int d = 0; d < 128; ++d) { float t = col[d * 68]; a += t * t; }
        red[q * 64 + k] = a;
    }
    __syncthreads();
    if (tid < 64) scal[tid] = red[tid] + red[64 + tid] + red[128 + tid] + red[192 + tid];
    __syncthreads();
    if (tid == 0) {
        float g = 0.0f;
        for (int i = 0; i < 64; ++i) { float ss = scal[i]; g += ss / (ss + EPSF); }
        gsp[0] = rsqrtf(g + EPSF);
    }
    __syncthreads();
    if (tid < 64) scal[tid] = rsqrtf(scal[tid] + EPSF);
    __syncthreads();

    const float g = gsp[0];
    #pragma unroll 4
    for (int pass = 0; pass < 32; ++pass) {
        const int idx = pass * 256 + tid;
        const int d = idx >> 4, k4 = (idx & 15) * 4;
        float4 val = *(const float4*)(vb + d * 68 + k4);
        val.x *= scal[k4 + 0] * g;
        val.y *= scal[k4 + 1] * g;
        val.z *= scal[k4 + 2] * g;
        val.w *= scal[k4 + 3] * g;
        *(float4*)(vn + d * 64 + k4) = val;
    }
}

// =====================================================================
extern "C" void kernel_launch(void* const* d_in, const int* in_sizes, int n_in,
                              void* d_out, int out_size, void* d_ws, size_t ws_size,
                              hipStream_t stream)
{
    const float* x    = (const float*)d_in[0];   // [64,32,32,512]
    const float* W    = (const float*)d_in[1];   // [512,64]
    const float* bias = (const float*)d_in[2];   // [64]
    const float* Cmat = (const float*)d_in[3];   // [512,64]
    float* out = (float*)d_out;                  // [64, 512*64]

    unsigned short* a_ws = (unsigned short*)d_ws;                       // bf16 [65536][64]
    float* a_sum = (float*)((char*)d_ws + (size_t)PTOT * KCL * 2);      // fp32 [64][64]

    hipMemsetAsync(a_sum, 0, NIMG * KCL * sizeof(float), stream);

    k_assign_softmax<<<PTOT / 128, 256, 0, stream>>>(x, W, bias, a_ws, a_sum);
    k_vlad_gemm<<<NIMG * 4, 256, 0, stream>>>(x, a_ws, a_sum, Cmat, out);

    const size_t smem3 = (size_t)(512 * 68 + 256 + 64 + 4) * sizeof(float);
    k_normalize<<<NIMG, 256, smem3, stream>>>(out);
}
